// SimVQ_76922864271853
// MI455X (gfx1250) — matrix operand order repaired and checked by reading, not compiled, not run on hardware
//
#include <hip/hip_runtime.h>
#include <stdint.h>

// SimVQ on MI455X (gfx1250, wave32).
//  1) proj_norm_wmma (f32 WMMA 16x16x4): codebook = E @ We^T + be -> raw f32 + l2-norm f16
//  2) proj_norm_wmma:                    z_n = normalize(z @ Wz^T + bz) -> f16
//  3) dist_argmin (f16 WMMA 16x16x32, K=512): fused argmax_n(z_n . c_n),
//     codebook tiles streamed into LDS with GLOBAL_LOAD_ASYNC_TO_LDS_B128
//     (ASYNCcnt), double-buffered, overlapped with chained WMMAs
//  4) gather + STE output + deterministic partial loss sums
//  5) finalize loss = 1.25 * mean((q-z)^2)

#define D_DIM   512
#define N_EMB   8192
#define BT      32768              // 8*4096
#define Q_ELEMS (BT * D_DIM)       // 16777216

typedef __attribute__((ext_vector_type(16))) _Float16 v16h;
typedef __attribute__((ext_vector_type(8)))  _Float16 v8h;
typedef __attribute__((ext_vector_type(8)))  float    v8f;
typedef __attribute__((ext_vector_type(2)))  float    v2f;

union V16U { v16h v; v8h h[2]; };

// ---------------------------------------------------------------------------
// Projection + row L2-norm via V_WMMA_F32_16X16X4_F32.
// Block = 8 waves = one 16-row tile of X; wave w owns 4 column tiles
// (cols t*128 + w*16). A frag: lane holds X[r][k0+2*sel .. +2] (float2).
// B frag: lane holds W[c][k0+2*sel .. +2]. 128 K-steps, A shared across the
// 4 accumulators. Result staged in LDS for the row norm.
// ---------------------------------------------------------------------------
__global__ void __launch_bounds__(256)
proj_norm_wmma(const float* __restrict__ X, const float* __restrict__ W,
               const float* __restrict__ bias,
               float* __restrict__ out_raw, _Float16* __restrict__ out_h) {
    __shared__ float sP[16 * D_DIM];      // 32 KB
    __shared__ float sRed[16 * 16];
    __shared__ float sInv[16];

    const int tid  = threadIdx.x;
    const int wave = tid >> 5;
    const int lane = tid & 31;
    const int sel  = lane >> 4;
    const int m0   = blockIdx.x * 16;
    const int r    = m0 + (lane & 15);

    const float* xrow = X + (size_t)r * D_DIM + 2 * sel;

    int cols[4];
    const float* wrow[4];
#pragma unroll
    for (int t = 0; t < 4; ++t) {
        cols[t] = t * 128 + wave * 16 + (lane & 15);
        wrow[t] = W + (size_t)cols[t] * D_DIM + 2 * sel;
    }

    v8f acc[4] = {v8f{}, v8f{}, v8f{}, v8f{}};
#pragma unroll 8
    for (int s = 0; s < D_DIM / 4; ++s) {
        const v2f a = *(const v2f*)(xrow + s * 4);
#pragma unroll
        for (int t = 0; t < 4; ++t) {
            const v2f b = *(const v2f*)(wrow[t] + s * 4);
            acc[t] = __builtin_amdgcn_wmma_f32_16x16x4_f32(
                false, a, false, b, (short)0, acc[t], false, false);
        }
    }

    // Add bias, park C tiles in LDS. C layout: lane n=(lane&15) -> col,
    // VGPR j -> row j + 8*sel.
#pragma unroll
    for (int t = 0; t < 4; ++t) {
        const float bv = bias[cols[t]];
        const int   cl = cols[t];
#pragma unroll
        for (int j = 0; j < 8; ++j)
            sP[(j + 8 * sel) * D_DIM + cl] = acc[t][j] + bv;
    }
    __syncthreads();

    {   // row sum of squares: 16 threads per row, 32 elems each
        const int row = tid >> 4, part = tid & 15;
        const float* p = sP + row * D_DIM + part * 32;
        float s = 0.f;
#pragma unroll
        for (int i = 0; i < 32; ++i) s += p[i] * p[i];
        sRed[row * 16 + part] = s;
    }
    __syncthreads();
    if (tid < 16) {
        float s = 0.f;
#pragma unroll
        for (int i = 0; i < 16; ++i) s += sRed[tid * 16 + i];
        sInv[tid] = 1.0f / fmaxf(sqrtf(s), 1e-12f);
    }
    __syncthreads();

#pragma unroll
    for (int i = 0; i < 32; ++i) {
        const int lin = i * 256 + tid;
        const int rr = lin >> 9, cc = lin & 511;
        const float v = sP[rr * D_DIM + cc];
        const size_t e = (size_t)(m0 + rr) * D_DIM + cc;
        if (out_raw) out_raw[e] = v;
        out_h[e] = (_Float16)(v * sInv[rr]);
    }
}

// ---------------------------------------------------------------------------
// Fused cosine-similarity GEMM + argmax, codebook tiles streamed into LDS
// with the CDNA5 async data mover (GLOBAL_LOAD_ASYNC_TO_LDS_B128, ASYNCcnt).
// Block = 8 waves; wave w owns rows m0 = blockIdx*128 + w*16 with its full
// K=512 A fragment set preloaded (128 VGPRs). The block sweeps all 8192
// codes in 32-column tiles, double-buffered in LDS:
//   s_wait_asynccnt 0 ; barrier ; issue async loads of tile t+1 into buf^1 ;
//   2x16 chained v_wmma_f32_16x16x32_f16 on tile t from buf.
// LDS row stride 520 halfs -> 16-lane B reads are bank-conflict free.
// ---------------------------------------------------------------------------
#define TILE_COLS 32
#define LROW      (D_DIM + 8)                 // padded stride in halfs
#define BUF_HALFS (TILE_COLS * LROW)          // 16,640 halfs = 33,280 B

// One b128 async copy: per-lane LDS byte address in a VGPR, 64-bit global
// address in a VGPR pair, literal instruction offset.
#define ASYNC_LD16(LDSOFF, GADDR, IMM)                                        \
    asm volatile("global_load_async_to_lds_b128 %0, %1, off offset:" #IMM    \
                 :: "v"(LDSOFF), "v"(GADDR) : "memory")

__global__ void __launch_bounds__(256)
dist_argmin_kernel(const _Float16* __restrict__ zh, const _Float16* __restrict__ ch,
                   int* __restrict__ idx_out, float* __restrict__ idx_out_f) {
    __shared__ _Float16 sbuf[2 * BUF_HALFS];  // 66,560 B double buffer

    const int tid  = threadIdx.x;
    const int wave = tid >> 5;
    const int lane = tid & 31;
    const int sel  = lane >> 4;
    const int m0   = blockIdx.x * 128 + wave * 16;
    const int r    = m0 + (lane & 15);

    // A fragments for all 16 K-steps: lane holds row r, K {sel*8..+8, 16+sel*8..+8}
    V16U a[16];
    const _Float16* zrow = zh + (size_t)r * D_DIM + sel * 8;
#pragma unroll
    for (int s = 0; s < 16; ++s) {
        a[s].h[0] = *(const v8h*)(zrow + s * 32);
        a[s].h[1] = *(const v8h*)(zrow + s * 32 + 16);
    }

    float bestv[8];
    int   besti[8];
#pragma unroll
    for (int j = 0; j < 8; ++j) { bestv[j] = -3.0e38f; besti[j] = 0; }

    // Async staging map: thread t moves 128 contiguous bytes (64 halfs) of
    // tile column ci = t>>3, segment seg = t&7, as 8 b128 transfers.
    const int ci = tid >> 3, seg = tid & 7;
    const _Float16* gsrc = ch + (size_t)ci * D_DIM + seg * 64;
    const uint32_t  ldsoff0 =
        (uint32_t)(uintptr_t)(&sbuf[ci * LROW + seg * 64]);   // LDS byte addr, buf 0
    const uint32_t  ldsoff1 = ldsoff0 + BUF_HALFS * 2;        // buf 1

#define ISSUE_TILE(LDSO, GP)                                                  \
    do {                                                                      \
        ASYNC_LD16(LDSO, GP, 0);   ASYNC_LD16(LDSO, GP, 16);                  \
        ASYNC_LD16(LDSO, GP, 32);  ASYNC_LD16(LDSO, GP, 48);                  \
        ASYNC_LD16(LDSO, GP, 64);  ASYNC_LD16(LDSO, GP, 80);                  \
        ASYNC_LD16(LDSO, GP, 96);  ASYNC_LD16(LDSO, GP, 112);                 \
    } while (0)

    ISSUE_TILE(ldsoff0, gsrc);                 // prologue: tile 0 -> buf 0

    const int khalf = sel * 8;    // B: lane holds K {sel*8..+8, 16+sel*8..+8}
    const _Float16* lbase = sbuf + (lane & 15) * LROW + khalf;

    for (int t = 0; t < N_EMB / TILE_COLS; ++t) {
        asm volatile("s_wait_asynccnt 0x0" ::: "memory");   // this wave's copies landed
        __syncthreads();                                    // everyone's copies landed
        if (t + 1 < N_EMB / TILE_COLS) {                    // stream tile t+1 -> other buf
            const _Float16* g2 = gsrc + (size_t)(t + 1) * TILE_COLS * D_DIM;
            const uint32_t  lo = ((t + 1) & 1) ? ldsoff1 : ldsoff0;
            ISSUE_TILE(lo, g2);
        }

        const _Float16* lt = lbase + (t & 1) * BUF_HALFS;
#pragma unroll
        for (int sub = 0; sub < 2; ++sub) {
            const _Float16* lb = lt + sub * 16 * LROW;
            v8f acc = {};
#pragma unroll
            for (int s = 0; s < 16; ++s) {
                V16U b;
                b.h[0] = *(const v8h*)(lb + s * 32);
                b.h[1] = *(const v8h*)(lb + s * 32 + 16);
                acc = __builtin_amdgcn_wmma_f32_16x16x32_f16(
                    false, a[s].v, false, b.v, (short)0, acc, false, false);
            }
            const int col = t * TILE_COLS + sub * 16 + (lane & 15);
#pragma unroll
            for (int j = 0; j < 8; ++j) {
                const float v = acc[j];
                if (v > bestv[j]) { bestv[j] = v; besti[j] = col; }
            }
        }
    }

    // Per-row argmax across the 16 column lanes (ties -> lower index).
    // Waves own disjoint rows, so no cross-wave reduction is needed.
#pragma unroll
    for (int j = 0; j < 8; ++j) {
        float v  = bestv[j];
        int   bi = besti[j];
#pragma unroll
        for (int m = 1; m <= 8; m <<= 1) {
            const float ov = __shfl_xor(v, m, 32);
            const int   oi = __shfl_xor(bi, m, 32);
            if (ov > v || (ov == v && oi < bi)) { v = ov; bi = oi; }
        }
        if ((lane & 15) == 0) {
            const int row = m0 + j + 8 * sel;
            idx_out[row]   = bi;
            idx_out_f[row] = (float)bi;
        }
    }
}

// ---------------------------------------------------------------------------
// Gather raw codebook rows, STE output z + (q - z), deterministic partials.
// ---------------------------------------------------------------------------
__global__ void __launch_bounds__(256)
gather_loss_kernel(const float* __restrict__ z, const float* __restrict__ craw,
                   const int* __restrict__ idx, float* __restrict__ out_q,
                   float* __restrict__ partials) {
    __shared__ float red[256];
    const size_t base = (size_t)blockIdx.x * 256 + threadIdx.x;
    float s = 0.f;
#pragma unroll
    for (int i = 0; i < 16; ++i) {
        const size_t e  = base + (size_t)i * (4096u * 256u);
        const int    bt = (int)(e >> 9);
        const int    d  = (int)(e & 511);
        const float  q  = craw[(size_t)idx[bt] * D_DIM + d];
        const float  zi = z[e];
        const float  t  = q - zi;       // matches reference rounding
        out_q[e] = zi + t;              // STE: z + (q - z)
        s += t * t;
    }
    red[threadIdx.x] = s;
    __syncthreads();
    for (int t = 128; t > 0; t >>= 1) {
        if (threadIdx.x < t) red[threadIdx.x] += red[threadIdx.x + t];
        __syncthreads();
    }
    if (threadIdx.x == 0) partials[blockIdx.x] = red[0];
}

__global__ void __launch_bounds__(256)
finalize_loss_kernel(const float* __restrict__ partials, float* __restrict__ out_loss) {
    __shared__ float red[256];
    float s = 0.f;
#pragma unroll
    for (int i = 0; i < 16; ++i) s += partials[threadIdx.x + i * 256];
    red[threadIdx.x] = s;
    __syncthreads();
    for (int t = 128; t > 0; t >>= 1) {
        if (threadIdx.x < t) red[threadIdx.x] += red[threadIdx.x + t];
        __syncthreads();
    }
    if (threadIdx.x == 0)
        out_loss[0] = 1.25f * red[0] / (float)Q_ELEMS;   // commit + 0.25*codebook
}

// ---------------------------------------------------------------------------
extern "C" void kernel_launch(void* const* d_in, const int* in_sizes, int n_in,
                              void* d_out, int out_size, void* d_ws, size_t ws_size,
                              hipStream_t stream) {
    (void)in_sizes; (void)n_in; (void)out_size; (void)ws_size;
    const float* z          = (const float*)d_in[0];
    const float* embedding  = (const float*)d_in[1];
    const float* emb_proj_w = (const float*)d_in[2];
    const float* emb_proj_b = (const float*)d_in[3];
    const float* z_proj_w   = (const float*)d_in[4];
    const float* z_proj_b   = (const float*)d_in[5];
    // d_in[6] = l2_scale: positive scale does not change the argmin -> unused.

    char* ws = (char*)d_ws;
    float*    c_raw    = (float*)ws;                                   // 16 MB
    _Float16* c_h      = (_Float16*)(ws + (16u << 20));                //  8 MB
    _Float16* z_h      = (_Float16*)(ws + (24u << 20));                // 32 MB
    int*      idx      = (int*)(ws + (56u << 20));                     // 128 KB
    float*    partials = (float*)(ws + (56u << 20) + (1u << 17));      // 16 KB

    float* out      = (float*)d_out;
    float* out_q    = out;                    // [BT, D]
    float* out_loss = out + Q_ELEMS;          // scalar
    float* out_idx  = out + Q_ELEMS + 1;      // [BT] as f32

    proj_norm_wmma<<<N_EMB / 16, 256, 0, stream>>>(embedding, emb_proj_w, emb_proj_b,
                                                   c_raw, c_h);
    proj_norm_wmma<<<BT / 16, 256, 0, stream>>>(z, z_proj_w, z_proj_b,
                                                nullptr, z_h);
    dist_argmin_kernel<<<BT / 128, 256, 0, stream>>>(z_h, c_h, idx, out_idx);
    gather_loss_kernel<<<4096, 256, 0, stream>>>(z, c_raw, idx, out_q, partials);
    finalize_loss_kernel<<<1, 256, 0, stream>>>(partials, out_loss);
}
